// HybridGPT_85504208929350
// MI455X (gfx1250) — matrix operand order repaired and checked by reading, not compiled, not run on hardware
//
#include <hip/hip_runtime.h>
#include <math.h>

// ---------------------------------------------------------------------------
// Model constants (from the reference)
// ---------------------------------------------------------------------------
static constexpr int B_   = 2;
static constexpr int T_   = 1024;
static constexpr int D_   = 1024;
static constexpr int H_   = 16;
static constexpr int KV_  = 4;
static constexpr int HD_  = 64;
static constexpr int V_   = 32768;
static constexpr int DFF_ = 4096;
static constexpr int CK_  = 4;
static constexpr int M_   = B_ * T_;          // 2048 rows everywhere
static constexpr float EPS_ = 1e-6f;

// ---------------------------------------------------------------------------
// WMMA types / helpers (CDNA5 gfx1250, wave32)
// ---------------------------------------------------------------------------
typedef __attribute__((ext_vector_type(16))) __bf16 v16bf;
typedef __attribute__((ext_vector_type(8)))  __bf16 bf16x8;
typedef __attribute__((ext_vector_type(8)))  float  v8f;

__device__ __forceinline__ v8f wmma_bf16(v16bf a, v16bf b, v8f c) {
  // D = A(16x32 bf16) x B(32x16 bf16) + C(16x16 f32)
  return __builtin_amdgcn_wmma_f32_16x16x32_bf16(false, a, false, b,
                                                 (short)0, c, false, false);
}

__device__ __forceinline__ v8f v8f_zero() {
  v8f z = {0.f, 0.f, 0.f, 0.f, 0.f, 0.f, 0.f, 0.f};
  return z;
}

// CDNA5 async global->LDS copy (16 bytes per lane), tracked by ASYNCcnt.
__device__ __forceinline__ void async_copy_b128(unsigned int lds_byte_addr,
                                                const void* gaddr) {
  asm volatile("global_load_async_to_lds_b128 %0, %1, off"
               :: "v"(lds_byte_addr),
                  "v"((unsigned long long)(size_t)gaddr)
               : "memory");
}
__device__ __forceinline__ void wait_asynccnt0() {
  asm volatile("s_wait_asynccnt 0x0" ::: "memory");
}

enum { ACT_NONE = 0, ACT_RELU2 = 1, ACT_SILU = 2, ACT_SIGMOID = 3 };

// ---------------------------------------------------------------------------
// Generic GEMM: C[M,N] = act( A[M,K](bf16) * W[N,K](bf16)^T ) (+ residual)
// Workgroup = 256 threads (8 waves). WG tile 128(M) x 128(N).
// Wave (wm,wn): wm=wave&3 -> 32-row strip, wn=wave>>2 -> 64-col strip,
// 2x4 accumulator fragments (8 WMMAs per K-step).
// LDS double-buffered; tiles staged with global_load_async_to_lds_b128.
// ---------------------------------------------------------------------------
static constexpr int LDSS = 40;  // bf16 row stride in LDS (pad vs 32)

__global__ __launch_bounds__(256) void gemm_bf16_kernel(
    const __bf16* __restrict__ A, const __bf16* __restrict__ W,
    float* __restrict__ C, __bf16* __restrict__ Cbf,
    const float* __restrict__ Res,
    int M, int N, int K, int ldc, int act)
{
  __shared__ __bf16 sA[2][128 * LDSS];
  __shared__ __bf16 sW[2][128 * LDSS];

  const int tid  = threadIdx.x;
  const int lane = tid & 31;
  const int wave = tid >> 5;
  const int wm   = wave & 3;               // 0..3 -> 32-row strip
  const int wn   = wave >> 2;              // 0..1 -> 64-col strip
  const int fm   = lane & 15;
  const int hi   = lane >> 4;              // 0 / 1 lane-half

  const int m0 = blockIdx.y * 128;
  const int n0 = blockIdx.x * 128;

  // staging assignment: thread -> (row, 16-col half) of the 128x32 tiles
  const int stRow  = tid >> 1;             // 0..127
  const int stCol  = (tid & 1) * 16;       // 0 or 16 (bf16 elements)
  int wRowC = n0 + stRow;                  // clamp W row (cols >= N never stored)
  if (wRowC > N - 1) wRowC = N - 1;
  const __bf16* gA0 = A + (size_t)(m0 + stRow) * K + stCol;
  const __bf16* gW0 = W + (size_t)wRowC * K + stCol;

  auto stage = [&](int buf, int kk) {
    const __bf16* ga = gA0 + kk;
    const __bf16* gw = gW0 + kk;
    const unsigned int la =
        (unsigned int)(size_t)(&sA[buf][stRow * LDSS + stCol]);
    const unsigned int lw =
        (unsigned int)(size_t)(&sW[buf][stRow * LDSS + stCol]);
    async_copy_b128(la,      ga);
    async_copy_b128(la + 16, ga + 8);
    async_copy_b128(lw,      gw);
    async_copy_b128(lw + 16, gw + 8);
  };

  v8f acc[2][4];
  #pragma unroll
  for (int s = 0; s < 2; ++s)
    #pragma unroll
    for (int t = 0; t < 4; ++t) acc[s][t] = v8f_zero();

  int buf = 0;
  stage(buf, 0);
  wait_asynccnt0();
  __syncthreads();

  for (int kk = 0; kk < K; kk += 32) {
    const bool more = (kk + 32 < K);
    if (more) stage(buf ^ 1, kk + 32);     // overlap next tile with compute

    // A fragments (two 16-row sub-tiles of this wave's 32 rows)
    v16bf afrag[2];
    #pragma unroll
    for (int s = 0; s < 2; ++s) {
      const __bf16* ar = &sA[buf][(wm * 32 + s * 16 + fm) * LDSS];
      #pragma unroll
      for (int e = 0; e < 8; ++e) {
        afrag[s][e]     = ar[hi * 16 + e];
        afrag[s][8 + e] = ar[hi * 16 + 8 + e];
      }
    }
    #pragma unroll
    for (int t = 0; t < 4; ++t) {
      v16bf bfrag;                         // n = fm, K contiguous 16 from hi*16
      const __bf16* br = &sW[buf][(wn * 64 + t * 16 + fm) * LDSS + hi * 16];
      #pragma unroll
      for (int e = 0; e < 16; ++e) bfrag[e] = br[e];
      acc[0][t] = wmma_bf16(afrag[0], bfrag, acc[0][t]);
      acc[1][t] = wmma_bf16(afrag[1], bfrag, acc[1][t]);
    }

    if (more) {
      wait_asynccnt0();                    // own copies for next tile complete
      __syncthreads();                     // everyone done reading cur + nxt ready
      buf ^= 1;
    }
  }

  // epilogue: C row = sub*16 + r + 8*hi, col = fm
  #pragma unroll
  for (int s = 0; s < 2; ++s) {
    #pragma unroll
    for (int t = 0; t < 4; ++t) {
      #pragma unroll
      for (int r = 0; r < 8; ++r) {
        const int row = m0 + wm * 32 + s * 16 + hi * 8 + r;
        const int col = n0 + wn * 64 + t * 16 + fm;
        if (col < N) {
          float v = acc[s][t][r];
          if      (act == ACT_RELU2)   { v = fmaxf(v, 0.f); v *= v; }
          else if (act == ACT_SILU)    { v = v / (1.f + __expf(-v)); }
          else if (act == ACT_SIGMOID) { v = 1.f / (1.f + __expf(-v)); }
          const size_t ci = (size_t)row * ldc + col;
          if (Res) v += Res[ci];
          if (C)   C[ci]   = v;
          if (Cbf) Cbf[ci] = (__bf16)v;
        }
      }
    }
  }
}

// ---------------------------------------------------------------------------
// Causal attention, one wave per (b, head, 16-query-row block).
// Scores via 2 WMMAs over HD=64; bf16 probs in LDS; P x V via WMMA.
// q layout [M, H*HD] (f32), k/v layout [M, KV*HD] (f32); out bf16 [M, H*HD].
// ---------------------------------------------------------------------------
__global__ __launch_bounds__(32) void attn_kernel(
    const float* __restrict__ q, const float* __restrict__ k,
    const float* __restrict__ v, __bf16* __restrict__ o)
{
  constexpr int PSTR = 1040;               // bf16 row stride (>= T + pad)
  __shared__ __bf16 sp[16 * PSTR];
  __shared__ float  rowsum[16];

  const int qb   = blockIdx.x;             // 0..T/16-1
  const int head = blockIdx.y;
  const int b    = blockIdx.z;
  const int lane = threadIdx.x;
  const int fm   = lane & 15;
  const int hi   = lane >> 4;
  const int kvh  = head / (H_ / KV_);

  // Q fragments (HD=64 split into two K=32 chunks)
  v16bf aq0, aq1;
  {
    const int tq = qb * 16 + fm;
    const float* qr = q + (size_t)(b * T_ + tq) * (H_ * HD_) + head * HD_;
    #pragma unroll
    for (int e = 0; e < 8; ++e) {
      aq0[e]     = (__bf16)qr[hi * 16 + e];
      aq0[8 + e] = (__bf16)qr[hi * 16 + 8 + e];
      aq1[e]     = (__bf16)qr[32 + hi * 16 + e];
      aq1[8 + e] = (__bf16)qr[32 + hi * 16 + 8 + e];
    }
  }

  const int ntiles = qb + 1;               // causal: k tiles 0..qb
  for (int j = 0; j < ntiles; ++j) {
    const int tk = j * 16 + fm;
    const float* kr = k + (size_t)(b * T_ + tk) * (KV_ * HD_) + kvh * HD_;
    v16bf bk0, bk1;                        // B frag: n=fm(time col), K=hd
    #pragma unroll
    for (int e = 0; e < 16; ++e) {
      bk0[e] = (__bf16)kr[hi * 16 + e];
      bk1[e] = (__bf16)kr[32 + hi * 16 + e];
    }
    v8f s = v8f_zero();
    s = wmma_bf16(aq0, bk0, s);
    s = wmma_bf16(aq1, bk1, s);
    #pragma unroll
    for (int r = 0; r < 8; ++r) {
      const int lr = hi * 8 + r;
      const float val = (j * 16 + fm <= qb * 16 + lr) ? s[r] : -1e30f;
      sp[lr * PSTR + j * 16 + fm] = (__bf16)val;
    }
  }

  const int ncols = ntiles * 16;
  const int ncpad = (ncols + 31) & ~31;
  if (ncpad != ncols) {                    // zero-pad 16 cols so P x V steps by 32
    for (int idx = lane; idx < 256; idx += 32) {
      const int rr = idx >> 4, cc = ncols + (idx & 15);
      sp[rr * PSTR + cc] = (__bf16)0.f;
    }
  }
  __syncthreads();

  // softmax: lane handles row fm, column parity hi
  {
    float mx = -1e30f;
    for (int c = hi; c < ncols; c += 2)
      mx = fmaxf(mx, (float)sp[fm * PSTR + c]);
    mx = fmaxf(mx, __shfl_xor(mx, 16, 32));
    float sum = 0.f;
    for (int c = hi; c < ncols; c += 2) {
      const float e = __expf((float)sp[fm * PSTR + c] - mx);
      sp[fm * PSTR + c] = (__bf16)e;
      sum += e;
    }
    sum += __shfl_xor(sum, 16, 32);
    if (hi == 0) rowsum[fm] = sum;
  }
  __syncthreads();

  // O = P x V
  v8f accO[4];
  #pragma unroll
  for (int t = 0; t < 4; ++t) accO[t] = v8f_zero();
  const int nch = ncpad >> 5;
  for (int ch = 0; ch < nch; ++ch) {
    v16bf ap;
    const __bf16* pr = sp + fm * PSTR + ch * 32;
    #pragma unroll
    for (int e = 0; e < 8; ++e) {
      ap[e]     = pr[hi * 16 + e];
      ap[8 + e] = pr[hi * 16 + 8 + e];
    }
    #pragma unroll
    for (int nt = 0; nt < 4; ++nt) {
      v16bf bv;                            // n=fm(out hd), K=time
      #pragma unroll
      for (int e = 0; e < 16; ++e) {
        const int tv = ch * 32 + hi * 16 + e;
        bv[e] = (__bf16)v[(size_t)(b * T_ + tv) * (KV_ * HD_) + kvh * HD_ + nt * 16 + fm];
      }
      accO[nt] = wmma_bf16(ap, bv, accO[nt]);
    }
  }

  #pragma unroll
  for (int nt = 0; nt < 4; ++nt) {
    #pragma unroll
    for (int r = 0; r < 8; ++r) {
      const int lr = hi * 8 + r;
      const int tq = qb * 16 + lr;
      o[(size_t)(b * T_ + tq) * (H_ * HD_) + head * HD_ + nt * 16 + fm] =
          (__bf16)(accO[nt][r] / rowsum[lr]);
    }
  }
}

// ---------------------------------------------------------------------------
// RMSNorm over D=1024 per row -> bf16
// ---------------------------------------------------------------------------
__global__ __launch_bounds__(256) void rmsn_kernel(
    const float* __restrict__ x, __bf16* __restrict__ out)
{
  __shared__ float red[256];
  const int row = blockIdx.x;
  const int tid = threadIdx.x;
  const float* xr = x + (size_t)row * D_;
  float ss = 0.f;
  for (int c = tid; c < D_; c += 256) { const float v = xr[c]; ss += v * v; }
  red[tid] = ss;
  __syncthreads();
  for (int s = 128; s > 0; s >>= 1) {
    if (tid < s) red[tid] += red[tid + s];
    __syncthreads();
  }
  const float scale = rsqrtf(red[0] / (float)D_ + EPS_);
  __bf16* orow = out + (size_t)row * D_;
  for (int c = tid; c < D_; c += 256) orow[c] = (__bf16)(xr[c] * scale);
}

// ---------------------------------------------------------------------------
// Per-head L2 norm (+optional gain, +optional RoPE). One 32-lane group per
// (row, head): lane holds elements lane and lane+32 (the rope pair).
// ---------------------------------------------------------------------------
__global__ __launch_bounds__(256) void l2n_rope_kernel(
    const float* __restrict__ in, int ldin,
    float* __restrict__ out, int ldout,
    const float* __restrict__ gain, int nheads, int use_rope)
{
  const int grp = threadIdx.x >> 5, lane = threadIdx.x & 31;
  const int gid = blockIdx.x * 8 + grp;
  const int total = B_ * T_ * nheads;
  if (gid >= total) return;
  const int h = gid % nheads;
  const int row = gid / nheads;
  const int t = row % T_;
  const float* p = in + (size_t)row * ldin + h * HD_;
  float x1 = p[lane], x2 = p[lane + 32];
  float ss = x1 * x1 + x2 * x2;
  #pragma unroll
  for (int off = 16; off >= 1; off >>= 1) ss += __shfl_xor(ss, off, 32);
  const float nrm = fmaxf(sqrtf(ss), EPS_);
  x1 /= nrm; x2 /= nrm;
  if (gain) { const float gn = gain[h]; x1 *= gn; x2 *= gn; }
  if (use_rope) {
    const float inv = __expf(-((float)lane / 32.f) * 9.2103403719761836f); // ln 1e4
    const float fr = (float)t * inv;
    const float c = cosf(fr), s = sinf(fr);
    const float o1 =  x1 * c + x2 * s;
    const float o2 = -x1 * s + x2 * c;
    x1 = o1; x2 = o2;
  }
  float* po = out + (size_t)row * ldout + h * HD_;
  po[lane] = x1; po[lane + 32] = x2;
}

// ---------------------------------------------------------------------------
// GDN: rmsn(o per-head 64) * g -> bf16
// ---------------------------------------------------------------------------
__global__ __launch_bounds__(256) void rms_gate_kernel(
    const float* __restrict__ o, const float* __restrict__ g,
    __bf16* __restrict__ out)
{
  const int grp = threadIdx.x >> 5, lane = threadIdx.x & 31;
  const int gid = blockIdx.x * 8 + grp;
  const int total = B_ * T_ * H_;
  if (gid >= total) return;
  const int h = gid % H_, row = gid / H_;
  const size_t base = (size_t)row * D_ + h * HD_;
  const float x1 = o[base + lane], x2 = o[base + lane + 32];
  float ss = x1 * x1 + x2 * x2;
  #pragma unroll
  for (int off = 16; off >= 1; off >>= 1) ss += __shfl_xor(ss, off, 32);
  const float sc = rsqrtf(ss / (float)HD_ + EPS_);
  out[base + lane]      = (__bf16)(x1 * sc * g[base + lane]);
  out[base + lane + 32] = (__bf16)(x2 * sc * g[base + lane + 32]);
}

// ---------------------------------------------------------------------------
// Causal depthwise conv (K=4) + SiLU over [M, 3D]
// ---------------------------------------------------------------------------
__global__ void dwconv_silu_kernel(
    const float* __restrict__ x, const float* __restrict__ w,
    float* __restrict__ out)
{
  const int n = B_ * T_ * 3 * D_;
  for (int idx = blockIdx.x * blockDim.x + threadIdx.x; idx < n;
       idx += gridDim.x * blockDim.x) {
    const int c = idx % (3 * D_);
    const int row = idx / (3 * D_);
    const int t = row % T_;
    float acc = 0.f;
    #pragma unroll
    for (int j = 0; j < CK_; ++j) {
      const int tt = t - (CK_ - 1) + j;
      if (tt >= 0) acc += x[(size_t)(row - t + tt) * (3 * D_) + c] * w[c * CK_ + j];
    }
    out[idx] = acc / (1.f + __expf(-acc));   // silu
  }
}

// ---------------------------------------------------------------------------
// Gated delta-net scan: one block per (head, batch); lane owns state column v,
// S[:,v] (64 floats) kept in registers; k/q staged through LDS each step.
// ---------------------------------------------------------------------------
__global__ __launch_bounds__(64) void gdn_scan_kernel(
    const float* __restrict__ q, const float* __restrict__ k,
    const float* __restrict__ v, int ldv, int vofs,
    const float* __restrict__ alpha, const float* __restrict__ beta,
    float* __restrict__ out)
{
  __shared__ float kk[HD_], qq[HD_];
  const int h = blockIdx.x, b = blockIdx.y;
  const int vc = threadIdx.x;
  float S[HD_];
  #pragma unroll
  for (int d = 0; d < HD_; ++d) S[d] = 0.f;

  for (int t = 0; t < T_; ++t) {
    const int row = b * T_ + t;
    kk[vc] = k[(size_t)row * D_ + h * HD_ + vc];
    qq[vc] = q[(size_t)row * D_ + h * HD_ + vc];
    __syncthreads();
    const float at = alpha[row * H_ + h];
    const float bt = beta[row * H_ + h];
    const float vt = v[(size_t)row * ldv + vofs + h * HD_ + vc];
    float kS = 0.f;
    #pragma unroll
    for (int d = 0; d < HD_; ++d) kS += kk[d] * S[d];
    float ot = 0.f;
    #pragma unroll
    for (int d = 0; d < HD_; ++d) {
      const float kd = kk[d];
      const float sv = at * (S[d] - bt * kd * kS) + kd * vt;
      S[d] = sv;
      ot += sv * qq[d];
    }
    out[(size_t)row * D_ + h * HD_ + vc] = ot;
    __syncthreads();
  }
}

// ---------------------------------------------------------------------------
// misc elementwise
// ---------------------------------------------------------------------------
__global__ void embed_kernel(const int* __restrict__ tok,
                             const float* __restrict__ emb,
                             float* __restrict__ x)
{
  const int n = M_ * D_;
  for (int idx = blockIdx.x * blockDim.x + threadIdx.x; idx < n;
       idx += gridDim.x * blockDim.x) {
    const int row = idx / D_, c = idx % D_;
    x[idx] = emb[(size_t)tok[row] * D_ + c];
  }
}

__global__ void f32_to_bf16_kernel(const float* __restrict__ in,
                                   __bf16* __restrict__ out, int n)
{
  for (int idx = blockIdx.x * blockDim.x + threadIdx.x; idx < n;
       idx += gridDim.x * blockDim.x)
    out[idx] = (__bf16)in[idx];
}

// ---------------------------------------------------------------------------
// Host orchestration
// ---------------------------------------------------------------------------
static inline void launch_cvt(hipStream_t s, const float* in, __bf16* out, size_t n) {
  int blocks = (int)((n + 255) / 256);
  if (blocks > 32768) blocks = 32768;
  f32_to_bf16_kernel<<<blocks, 256, 0, s>>>(in, out, (int)n);
}

static inline void launch_gemm(hipStream_t s, const __bf16* A, const __bf16* W,
                               float* C, __bf16* Cbf, const float* Res,
                               int N, int K, int ldc, int act) {
  dim3 grid((N + 127) / 128, M_ / 128);
  gemm_bf16_kernel<<<grid, 256, 0, s>>>(A, W, C, Cbf, Res, M_, N, K, ldc, act);
}

extern "C" void kernel_launch(void* const* d_in, const int* in_sizes, int n_in,
                              void* d_out, int out_size, void* d_ws, size_t ws_size,
                              hipStream_t stream) {
  (void)in_sizes; (void)n_in; (void)out_size; (void)ws_size;

  const int*   tokens = (const int*)  d_in[0];
  const float* emb    = (const float*)d_in[1];
  const float* aw_q   = (const float*)d_in[2];
  const float* aw_k   = (const float*)d_in[3];
  const float* aw_v   = (const float*)d_in[4];
  const float* aw_o   = (const float*)d_in[5];
  const float* a_gain = (const float*)d_in[6];
  const float* a_m1   = (const float*)d_in[7];
  const float* a_m2   = (const float*)d_in[8];
  const float* g_wq   = (const float*)d_in[9];
  const float* g_wk   = (const float*)d_in[10];
  const float* g_wv   = (const float*)d_in[11];
  const float* g_wa   = (const float*)d_in[12];
  const float* g_wb   = (const float*)d_in[13];
  const float* g_wg   = (const float*)d_in[14];
  const float* g_wo   = (const float*)d_in[15];
  const float* g_conv = (const float*)d_in[16];
  const float* g_m1   = (const float*)d_in[17];
  const float* g_m2   = (const float*)d_in[18];

  size_t off = 0;
  auto alloc = [&](size_t bytes) -> void* {
    off = (off + 255) & ~(size_t)255;
    void* p = (char*)d_ws + off;
    off += bytes;
    return p;
  };

  // bf16 weight copies (converted every launch; deterministic)
  const size_t nDD  = (size_t)D_ * D_;           // 1M
  const size_t nKVD = (size_t)KV_ * HD_ * D_;    // 256K
  const size_t nFD  = (size_t)DFF_ * D_;         // 4M
  __bf16* awqb = (__bf16*)alloc(2 * nDD  * 2);
  __bf16* awkb = (__bf16*)alloc(2 * nKVD * 2);
  __bf16* awvb = (__bf16*)alloc(2 * nKVD * 2);
  __bf16* awob = (__bf16*)alloc(2 * nDD  * 2);
  __bf16* am1b = (__bf16*)alloc(2 * nFD  * 2);
  __bf16* am2b = (__bf16*)alloc(2 * nFD  * 2);
  __bf16* gwqb = (__bf16*)alloc(2 * nDD  * 2);
  __bf16* gwkb = (__bf16*)alloc(2 * nDD  * 2);
  __bf16* gwvb = (__bf16*)alloc(2 * nDD  * 2);
  __bf16* gwab = (__bf16*)alloc(2 * (size_t)H_ * D_ * 2);
  __bf16* gwbb = (__bf16*)alloc(2 * (size_t)H_ * D_ * 2);
  __bf16* gwgb = (__bf16*)alloc(2 * nDD  * 2);
  __bf16* gwob = (__bf16*)alloc(2 * nDD  * 2);
  __bf16* gm1b = (__bf16*)alloc(2 * nFD  * 2);
  __bf16* gm2b = (__bf16*)alloc(2 * nFD  * 2);
  __bf16* embb = (__bf16*)alloc((size_t)V_ * D_ * 2);

  // activations
  float*  xf    = (float*) alloc((size_t)M_ * D_ * 4);
  __bf16* hbf   = (__bf16*)alloc((size_t)M_ * D_ * 2);
  float*  qf    = (float*) alloc((size_t)M_ * D_ * 4);
  float*  kf    = (float*) alloc((size_t)M_ * KV_ * HD_ * 4);
  float*  vf    = (float*) alloc((size_t)M_ * KV_ * HD_ * 4);
  __bf16* obf   = (__bf16*)alloc((size_t)M_ * D_ * 2);
  __bf16* midbf = (__bf16*)alloc((size_t)M_ * DFF_ * 2);
  float*  qkvf  = (float*) alloc((size_t)M_ * 3 * D_ * 4);
  float*  qkvc  = (float*) alloc((size_t)M_ * 3 * D_ * 4);
  float*  qg    = (float*) alloc((size_t)M_ * D_ * 4);
  float*  kg    = (float*) alloc((size_t)M_ * D_ * 4);
  float*  af    = (float*) alloc((size_t)M_ * H_ * 4);
  float*  bfb   = (float*) alloc((size_t)M_ * H_ * 4);
  float*  gof   = (float*) alloc((size_t)M_ * D_ * 4);
  float*  gf    = (float*) alloc((size_t)M_ * D_ * 4);
  __bf16* ogbf  = (__bf16*)alloc((size_t)M_ * D_ * 2);

  // weight conversions
  launch_cvt(stream, aw_q, awqb, 2 * nDD);
  launch_cvt(stream, aw_k, awkb, 2 * nKVD);
  launch_cvt(stream, aw_v, awvb, 2 * nKVD);
  launch_cvt(stream, aw_o, awob, 2 * nDD);
  launch_cvt(stream, a_m1, am1b, 2 * nFD);
  launch_cvt(stream, a_m2, am2b, 2 * nFD);
  launch_cvt(stream, g_wq, gwqb, 2 * nDD);
  launch_cvt(stream, g_wk, gwkb, 2 * nDD);
  launch_cvt(stream, g_wv, gwvb, 2 * nDD);
  launch_cvt(stream, g_wa, gwab, 2 * (size_t)H_ * D_);
  launch_cvt(stream, g_wb, gwbb, 2 * (size_t)H_ * D_);
  launch_cvt(stream, g_wg, gwgb, 2 * nDD);
  launch_cvt(stream, g_wo, gwob, 2 * nDD);
  launch_cvt(stream, g_m1, gm1b, 2 * nFD);
  launch_cvt(stream, g_m2, gm2b, 2 * nFD);
  launch_cvt(stream, emb,  embb, (size_t)V_ * D_);

  // embedding gather
  embed_kernel<<<8192, 256, 0, stream>>>(tokens, emb, xf);

  for (int layer = 0; layer < 2; ++layer) {
    // ---------------- attention block ----------------
    rmsn_kernel<<<M_, 256, 0, stream>>>(xf, hbf);
    launch_gemm(stream, hbf, awqb + layer * nDD,  qf, nullptr, nullptr, D_, D_, D_, ACT_NONE);
    launch_gemm(stream, hbf, awkb + layer * nKVD, kf, nullptr, nullptr, KV_ * HD_, D_, KV_ * HD_, ACT_NONE);
    launch_gemm(stream, hbf, awvb + layer * nKVD, vf, nullptr, nullptr, KV_ * HD_, D_, KV_ * HD_, ACT_NONE);
    l2n_rope_kernel<<<(M_ * H_) / 8, 256, 0, stream>>>(qf, D_, qf, D_,
        a_gain + layer * H_, H_, 1);
    l2n_rope_kernel<<<(M_ * KV_) / 8, 256, 0, stream>>>(kf, KV_ * HD_, kf, KV_ * HD_,
        (const float*)nullptr, KV_, 1);
    attn_kernel<<<dim3(T_ / 16, H_, B_), 32, 0, stream>>>(qf, kf, vf, obf);
    launch_gemm(stream, obf, awob + layer * nDD, xf, nullptr, xf, D_, D_, D_, ACT_NONE);
    rmsn_kernel<<<M_, 256, 0, stream>>>(xf, hbf);
    launch_gemm(stream, hbf, am1b + layer * nFD, nullptr, midbf, nullptr, DFF_, D_, DFF_, ACT_RELU2);
    launch_gemm(stream, midbf, am2b + layer * nFD, xf, nullptr, xf, D_, DFF_, D_, ACT_NONE);

    // ---------------- GDN block ----------------
    rmsn_kernel<<<M_, 256, 0, stream>>>(xf, hbf);
    launch_gemm(stream, hbf, gwqb + layer * nDD, qkvf,            nullptr, nullptr, D_, D_, 3 * D_, ACT_NONE);
    launch_gemm(stream, hbf, gwkb + layer * nDD, qkvf + D_,       nullptr, nullptr, D_, D_, 3 * D_, ACT_NONE);
    launch_gemm(stream, hbf, gwvb + layer * nDD, qkvf + 2 * D_,   nullptr, nullptr, D_, D_, 3 * D_, ACT_NONE);
    dwconv_silu_kernel<<<24576, 256, 0, stream>>>(qkvf, g_conv + layer * 3 * D_ * CK_, qkvc);
    l2n_rope_kernel<<<(M_ * H_) / 8, 256, 0, stream>>>(qkvc, 3 * D_, qg, D_,
        (const float*)nullptr, H_, 0);
    l2n_rope_kernel<<<(M_ * H_) / 8, 256, 0, stream>>>(qkvc + D_, 3 * D_, kg, D_,
        (const float*)nullptr, H_, 0);
    launch_gemm(stream, hbf, gwab + layer * (size_t)H_ * D_, af,  nullptr, nullptr, H_, D_, H_, ACT_SIGMOID);
    launch_gemm(stream, hbf, gwbb + layer * (size_t)H_ * D_, bfb, nullptr, nullptr, H_, D_, H_, ACT_SIGMOID);
    gdn_scan_kernel<<<dim3(H_, B_), HD_, 0, stream>>>(qg, kg, qkvc, 3 * D_, 2 * D_,
        af, bfb, gof);
    launch_gemm(stream, hbf, gwgb + layer * nDD, gf, nullptr, nullptr, D_, D_, D_, ACT_SILU);
    rms_gate_kernel<<<(M_ * H_) / 8, 256, 0, stream>>>(gof, gf, ogbf);
    launch_gemm(stream, ogbf, gwob + layer * nDD, xf, nullptr, xf, D_, D_, D_, ACT_NONE);
    rmsn_kernel<<<M_, 256, 0, stream>>>(xf, hbf);
    launch_gemm(stream, hbf, gm1b + layer * nFD, nullptr, midbf, nullptr, DFF_, D_, DFF_, ACT_RELU2);
    launch_gemm(stream, midbf, gm2b + layer * nFD, xf, nullptr, xf, D_, DFF_, D_, ACT_NONE);
  }

  // final logits: rmsn(x) @ emb^T -> d_out [M, V] f32
  rmsn_kernel<<<M_, 256, 0, stream>>>(xf, hbf);
  launch_gemm(stream, hbf, embb, (float*)d_out, nullptr, nullptr, V_, D_, V_, ACT_NONE);
}
